// LSTM_3719441678596
// MI455X (gfx1250) — matrix operand order repaired and checked by reading, not compiled, not run on hardware
//
#include <hip/hip_runtime.h>

// Fused 3-layer LSTM + output projection for MI455X (gfx1250, wave32, WMMA).
// One block = 16 batch rows x 10 waves (one wave per 16-gate tile of 4H=160).
// All recurrent state (h,c) lives in LDS for the whole T=512 loop; weights
// live in per-wave registers as WMMA B-fragments. HBM traffic ~= read x (25MB)
// + write out (4MB) -> near the 23.3 TB/s bandwidth floor for this problem.
// Activations use raw v_exp_f32/v_rcp_f32 (branch-free) since the elementwise
// stage sits on the T=512 sequential critical path.

typedef __attribute__((ext_vector_type(16))) _Float16 v16h;
typedef __attribute__((ext_vector_type(8)))  float    v8f;

#define B_   2048
#define T_   512
#define D_   6
#define H_   40
#define G_   160          // 4*H
#define ROWS 16           // batch rows per block (WMMA M)
#define NW   10           // waves per block: one per 16-gate tile (10*16 = 160)
#define NT   (NW * 32)    // 320 threads

#define LOG2E 1.44269504088896f

// Branch-free activations on hardware transcendentals (v_exp_f32, v_rcp_f32).
__device__ __forceinline__ float sigm(float v) {
  return __builtin_amdgcn_rcpf(1.f + __builtin_amdgcn_exp2f(-LOG2E * v));
}
__device__ __forceinline__ float tanh_fast(float v) {
  // tanh(x) = 1 - 2/(e^{2x}+1); saturates to +/-1 via exp2->inf/0, rcp(inf)=0.
  return 1.f - 2.f * __builtin_amdgcn_rcpf(1.f + __builtin_amdgcn_exp2f(2.f * LOG2E * v));
}

// ---- WMMA B fragment (32x16 f16): lane L holds N=gate0+(L&15), g=(L&31)>>4;
//      halves 0..7 -> K = base + 8g + i, halves 8..15 -> K = base + 16 + 8g + i.
//      B[k][n] = W[gate0+n][k]  (W is row-major [160][kdim] f32 in HBM).
__device__ __forceinline__ v16h load_bfrag(const float* __restrict__ W,
                                           int gate0, int kdim, int k) {
  v16h f;
  const int ln = k & 31, base = k - ln, n = ln & 15, h = ln >> 4;
#pragma unroll
  for (int s = 0; s < 16; ++s) {
    const int kk = base + (s < 8 ? 8 * h + s : 16 + 8 * h + (s - 8));
    f[s] = (kk < kdim) ? (_Float16)W[(gate0 + n) * kdim + kk] : (_Float16)0.f;
  }
  return f;
}

// ---- WMMA A fragment (16x32 f16) from LDS row-major [16][ld] buffer.
//      lane L: M=L&15, g=L>>4; halves 0..7 -> K = 32c + 8g + i,
//                              halves 8..15 -> K = 32c + 16 + 8g + i.
__device__ __forceinline__ v16h load_afrag(const _Float16* buf, int ld,
                                           int lane, int c) {
  const int m = lane & 15, g = lane >> 4;
  const _Float16* r0 = buf + m * ld + c * 32 + g * 8;
  v16h a;
#pragma unroll
  for (int i = 0; i < 8; ++i) { a[i] = r0[i]; a[8 + i] = r0[16 + i]; }
  return a;
}

// ---- One LSTM layer, one timestep, for this block's 16-row batch tile.
template <int IN_CHUNKS>
__device__ __forceinline__ void layer_step(
    const _Float16* inb, int in_ld,                 // input activations (x or prev h)
    _Float16 (&hb)[ROWS][64], float (&cb)[ROWS][H_],// this layer's recurrent state
    v16h wihA, v16h wihB, v16h whhA, v16h whhB,     // per-wave weight B-fragments
    float biasv, float (&gb)[ROWS][G_],
    int lane, int wv, int tid) {
  v8f acc;
#pragma unroll
  for (int i = 0; i < 8; ++i) acc[i] = biasv;      // C init = bih+bhh (per N col)

  {
    v16h a = load_afrag(inb, in_ld, lane, 0);
    acc = __builtin_amdgcn_wmma_f32_16x16x32_f16(false, a, false, wihA,
                                                 (short)0, acc, false, false);
  }
  if (IN_CHUNKS > 1) {
    v16h a = load_afrag(inb, in_ld, lane, 1);
    acc = __builtin_amdgcn_wmma_f32_16x16x32_f16(false, a, false, wihB,
                                                 (short)0, acc, false, false);
  }
  {
    v16h a = load_afrag(&hb[0][0], 64, lane, 0);
    acc = __builtin_amdgcn_wmma_f32_16x16x32_f16(false, a, false, whhA,
                                                 (short)0, acc, false, false);
    a = load_afrag(&hb[0][0], 64, lane, 1);
    acc = __builtin_amdgcn_wmma_f32_16x16x32_f16(false, a, false, whhB,
                                                 (short)0, acc, false, false);
  }

  // Scatter C (layout: VGPR v, lane L -> M = v + 8*(L>>4), N = L&15) to LDS gates.
  const int g = lane >> 4, m = lane & 15;
#pragma unroll
  for (int v = 0; v < 8; ++v) gb[v + 8 * g][16 * wv + m] = acc[v];
  __syncthreads();

  // Elementwise gate nonlinearity + state update: 16*40 = 640 elems / 320 thr.
#pragma unroll
  for (int q = 0; q < 2; ++q) {
    const int e = 2 * tid + q;                     // 0..639
    const int r = e / H_, j = e - r * H_;
    const float iv = sigm(gb[r][j]);
    const float fv = sigm(gb[r][H_ + j]);
    const float gv = tanh_fast(gb[r][2 * H_ + j]);
    const float ov = sigm(gb[r][3 * H_ + j]);
    const float cc = fv * cb[r][j] + iv * gv;
    cb[r][j] = cc;
    hb[r][j] = (_Float16)(ov * tanh_fast(cc));
  }
  __syncthreads();
}

__global__ __launch_bounds__(NT) void lstm3_fused(
    const float* __restrict__ x,
    const float* __restrict__ Wih0, const float* __restrict__ Whh0,
    const float* __restrict__ bih0, const float* __restrict__ bhh0,
    const float* __restrict__ Wih1, const float* __restrict__ Whh1,
    const float* __restrict__ bih1, const float* __restrict__ bhh1,
    const float* __restrict__ Wih2, const float* __restrict__ Whh2,
    const float* __restrict__ bih2, const float* __restrict__ bhh2,
    const float* __restrict__ Wout, const float* __restrict__ bout,
    float* __restrict__ out) {
  __shared__ _Float16 xb[ROWS][32];        // layer-0 input tile (K padded to 32)
  __shared__ _Float16 hb[3][ROWS][64];     // per-layer h (f16, K padded to 64)
  __shared__ float    cb[3][ROWS][H_];     // per-layer c
  __shared__ float    gb[ROWS][G_];        // gate pre-activations (f32)
  __shared__ float    woutL[H_];

  const int tid  = threadIdx.x;
  const int lane = tid & 31;
  const int wv   = tid >> 5;
  const int row0 = blockIdx.x * ROWS;

  // Zero-init state (h0 = c0 = 0; padding columns stay 0 forever).
  for (int i = tid; i < ROWS * 32;     i += NT) (&xb[0][0])[i]    = (_Float16)0.f;
  for (int i = tid; i < 3 * ROWS * 64; i += NT) (&hb[0][0][0])[i] = (_Float16)0.f;
  for (int i = tid; i < 3 * ROWS * H_; i += NT) (&cb[0][0][0])[i] = 0.f;
  if (tid < H_) woutL[tid] = Wout[tid];

  // Persistent per-wave weight fragments (reused for all 512 timesteps).
  const int gate0 = 16 * wv;
  const int m = lane & 15;
  v16h wih0A = load_bfrag(Wih0, gate0, D_, lane);            // K=6 fits chunk 0
  v16h whh0A = load_bfrag(Whh0, gate0, H_, lane);
  v16h whh0B = load_bfrag(Whh0, gate0, H_, 32 + lane);
  v16h wih1A = load_bfrag(Wih1, gate0, H_, lane);
  v16h wih1B = load_bfrag(Wih1, gate0, H_, 32 + lane);
  v16h whh1A = load_bfrag(Whh1, gate0, H_, lane);
  v16h whh1B = load_bfrag(Whh1, gate0, H_, 32 + lane);
  v16h wih2A = load_bfrag(Wih2, gate0, H_, lane);
  v16h wih2B = load_bfrag(Wih2, gate0, H_, 32 + lane);
  v16h whh2A = load_bfrag(Whh2, gate0, H_, lane);
  v16h whh2B = load_bfrag(Whh2, gate0, H_, 32 + lane);
  const float bias0 = bih0[gate0 + m] + bhh0[gate0 + m];
  const float bias1 = bih1[gate0 + m] + bhh1[gate0 + m];
  const float bias2 = bih2[gate0 + m] + bhh2[gate0 + m];
  const float boutv = bout[0];

  __syncthreads();

  for (int t = 0; t < T_; ++t) {
    // Stage this timestep's x tile (16 rows x 6 feats) into LDS as f16.
    if (tid < ROWS * D_) {
      const int r = tid / D_, d = tid - r * D_;
      xb[r][d] = (_Float16)x[((long)(row0 + r) * T_ + t) * D_ + d];
    }
    if (tid < ROWS && t + 1 < T_)
      __builtin_prefetch(&x[((long)(row0 + tid) * T_ + (t + 1)) * D_], 0, 0);
    __syncthreads();

    layer_step<1>(&xb[0][0],    32, hb[0], cb[0], wih0A, wih0A, whh0A, whh0B,
                  bias0, gb, lane, wv, tid);
    layer_step<2>(&hb[0][0][0], 64, hb[1], cb[1], wih1A, wih1B, whh1A, whh1B,
                  bias1, gb, lane, wv, tid);
    layer_step<2>(&hb[1][0][0], 64, hb[2], cb[2], wih2A, wih2B, whh2A, whh2B,
                  bias2, gb, lane, wv, tid);

    // Output projection: out[b,t] = h2[b,:] . Wout + bout
    if (tid < ROWS) {
      float s = boutv;
#pragma unroll
      for (int j = 0; j < H_; ++j) s += (float)hb[2][tid][j] * woutL[j];
      out[(long)(row0 + tid) * T_ + t] = s;
    }
  }
}

extern "C" void kernel_launch(void* const* d_in, const int* in_sizes, int n_in,
                              void* d_out, int out_size, void* d_ws, size_t ws_size,
                              hipStream_t stream) {
  (void)in_sizes; (void)n_in; (void)out_size; (void)d_ws; (void)ws_size;
  const float* xp    = (const float*)d_in[0];
  const float* Wih0  = (const float*)d_in[1];
  const float* Whh0  = (const float*)d_in[2];
  const float* bih0  = (const float*)d_in[3];
  const float* bhh0  = (const float*)d_in[4];
  const float* Wih1  = (const float*)d_in[5];
  const float* Whh1  = (const float*)d_in[6];
  const float* bih1  = (const float*)d_in[7];
  const float* bhh1  = (const float*)d_in[8];
  const float* Wih2  = (const float*)d_in[9];
  const float* Whh2  = (const float*)d_in[10];
  const float* bih2  = (const float*)d_in[11];
  const float* bhh2  = (const float*)d_in[12];
  const float* Wout  = (const float*)d_in[13];
  const float* bout  = (const float*)d_in[14];
  float* outp = (float*)d_out;

  dim3 grid(B_ / ROWS);   // 128 blocks of 16 batch rows
  dim3 block(NT);         // 10 waves: one per 16-gate tile
  lstm3_fused<<<grid, block, 0, stream>>>(
      xp, Wih0, Whh0, bih0, bhh0, Wih1, Whh1, bih1, bhh1,
      Wih2, Whh2, bih2, bhh2, Wout, bout, outp);
}
